// Encoder_74294344286564
// MI455X (gfx1250) — matrix operand order repaired and checked by reading, not compiled, not run on hardware
//
#include <hip/hip_runtime.h>
#include <hip/hip_bf16.h>
#include <math.h>

// ---- problem dims (fixed by reference) ----
#define VD   32000
#define DD   1024
#define HH_  16
#define FF   4096
#define LL   6
#define DK   64
#define BB   2
#define SS   2048
#define MR   (BB*SS)      // 4096 flattened activation rows

typedef __attribute__((ext_vector_type(16))) __bf16 v16bf;
typedef __attribute__((ext_vector_type(8)))  float  v8f;

union Frag16 { v16bf bf; unsigned int u[8]; };

__device__ __forceinline__ unsigned short f2bf(float x) {
    union { __bf16 b; unsigned short u; } c; c.b = (__bf16)x; return c.u;
}
__device__ __forceinline__ unsigned int pack2(float lo, float hi) {
    union { __bf16 b[2]; unsigned int u; } c;
    c.b[0] = (__bf16)lo; c.b[1] = (__bf16)hi; return c.u;
}

__device__ __forceinline__ v8f wmma_bf16(const Frag16& a, const Frag16& b, v8f c) {
    return __builtin_amdgcn_wmma_f32_16x16x32_bf16(false, a.bf, false, b.bf,
                                                   (short)0, c, false, false);
}

// ---- async global->LDS copy (gfx1250 ASYNCcnt path), guarded ----
#if defined(__has_builtin)
#if __has_builtin(__builtin_amdgcn_global_load_async_to_lds_b128) && \
    __has_builtin(__builtin_amdgcn_s_wait_asynccnt)
#define USE_ASYNC_LDS 1
#endif
#endif

#ifdef USE_ASYNC_LDS
typedef int v4i __attribute__((ext_vector_type(4)));
typedef __attribute__((address_space(1))) v4i* as1_v4i;   // global (__device__)
typedef __attribute__((address_space(3))) v4i* as3_v4i;   // LDS (__shared__)
#define CP16B(ldst, gsrc) \
    __builtin_amdgcn_global_load_async_to_lds_b128((as1_v4i)(gsrc), (as3_v4i)(ldst), 0, 0)
#define CP_WAIT() __builtin_amdgcn_s_wait_asynccnt(0)
#endif

// =====================================================================
// Embedding + (batch-indexed) positional encoding:  x = emb[src] + pe[b]
// =====================================================================
__global__ __launch_bounds__(256) void embed_pe_kernel(
    const int* __restrict__ src, const float* __restrict__ emb,
    float* __restrict__ x)
{
    int i = blockIdx.x * 256 + threadIdx.x;          // < BB*SS*DD
    int b = i / (SS * DD);
    int r = i - b * (SS * DD);
    int s = r / DD;
    int d = r - s * DD;
    int tok = src[b * SS + s];
    int i2 = d & ~1;
    float div = expf((float)i2 * (-logf(10000.f) / (float)DD));
    float arg = (float)b / div;
    float pe = (d & 1) ? cosf(arg) : sinf(arg);
    x[i] = emb[(size_t)tok * DD + d] + pe;
}

// =====================================================================
// GEMM: C = A[M,K] @ W[K,N] + bias (+res)(ReLU). bf16 WMMA core, f32 acc.
// 256 thr (8 waves), tile 128x128, K-step 32, double-buffered LDS.
// out_mode: 0 = f32, 1 = bf16 row-major, 2 = bf16 into V^T[B,H,dk,S]
// =====================================================================
#define TM 128
#define TN 128
#define KB 32

struct StageRegs { float4 ra[4]; float4 rb[2][2]; };

__device__ __forceinline__ void stage_load(const float* __restrict__ A,
                                           const float* __restrict__ W,
                                           int K, int N, int bm, int bn,
                                           int k0, int t, StageRegs& s)
{
    int a_cg = t & 7, a_row0 = t >> 3;
#pragma unroll
    for (int r = 0; r < 4; ++r) {
        int row = a_row0 + r * 32;
        s.ra[r] = *(const float4*)(A + (size_t)(bm + row) * K + k0 + a_cg * 4);
    }
    int b_cg = t & 31;
#pragma unroll
    for (int r = 0; r < 2; ++r) {
        int p = (t >> 5) + r * 8;            // k-pair index 0..15
        int kk = 2 * p;
        s.rb[r][0] = *(const float4*)(W + (size_t)(k0 + kk)     * N + bn + b_cg * 4);
        s.rb[r][1] = *(const float4*)(W + (size_t)(k0 + kk + 1) * N + bn + b_cg * 4);
    }
}

__device__ __forceinline__ void stage_store(unsigned int* As, unsigned int* Bs,
                                            int t, const StageRegs& s)
{
    int a_cg = t & 7, a_row0 = t >> 3;
#pragma unroll
    for (int r = 0; r < 4; ++r) {
        int row = a_row0 + r * 32;
        As[row * 16 + a_cg * 2 + 0] = pack2(s.ra[r].x, s.ra[r].y);
        As[row * 16 + a_cg * 2 + 1] = pack2(s.ra[r].z, s.ra[r].w);
    }
    int b_cg = t & 31;
#pragma unroll
    for (int r = 0; r < 2; ++r) {
        int p = (t >> 5) + r * 8;
        int col = b_cg * 4;
        float lo[4] = { s.rb[r][0].x, s.rb[r][0].y, s.rb[r][0].z, s.rb[r][0].w };
        float hi[4] = { s.rb[r][1].x, s.rb[r][1].y, s.rb[r][1].z, s.rb[r][1].w };
#pragma unroll
        for (int i = 0; i < 4; ++i)
            Bs[(col + i) * 16 + p] = pack2(lo[i], hi[i]);
    }
}

__global__ __launch_bounds__(256) void gemm_bf16_kernel(
    const float* __restrict__ A, const float* __restrict__ W,
    const float* __restrict__ bias, const float* __restrict__ res,
    void* __restrict__ Cout, int M, int N, int K, int relu, int out_mode)
{
    __shared__ unsigned int As[2][TM * KB / 2];   // [buf][row*16 + k/2]
    __shared__ unsigned int Bs[2][TN * KB / 2];   // [buf][col*16 + k/2]

    const int t    = threadIdx.x;
    const int wave = t >> 5;
    const int lane = t & 31;
    const int hh   = lane >> 4;
    const int r0   = lane & 15;
    const int wm   = wave & 3;
    const int wn   = wave >> 2;
    const int bm   = blockIdx.y * TM;
    const int bn   = blockIdx.x * TN;

    v8f acc[2][4];
#pragma unroll
    for (int mt = 0; mt < 2; ++mt)
#pragma unroll
        for (int nt = 0; nt < 4; ++nt)
#pragma unroll
            for (int j = 0; j < 8; ++j) acc[mt][nt][j] = 0.f;

    StageRegs sr;
    stage_load(A, W, K, N, bm, bn, 0, t, sr);
    stage_store(As[0], Bs[0], t, sr);
    __syncthreads();

    const int nsteps = K / KB;
    for (int it = 0; it < nsteps; ++it) {
        const int cur = it & 1;
        if (it + 1 < nsteps)
            stage_load(A, W, K, N, bm, bn, (it + 1) * KB, t, sr);   // overlap w/ WMMA
        if (it + 2 < nsteps) {   // global_prefetch_b8, WGP scope (locality=3)
            __builtin_prefetch(A + (size_t)(bm + (t >> 3)) * K + (it + 2) * KB, 0, 3);
            __builtin_prefetch(W + (size_t)((it + 2) * KB + (t >> 5)) * N + bn + (t & 31) * 4, 0, 3);
        }

        const unsigned short* Ab = (const unsigned short*)As[cur];
        const unsigned short* Bb = (const unsigned short*)Bs[cur];
        Frag16 af[2];
#pragma unroll
        for (int mt = 0; mt < 2; ++mt) {
            int row = wm * 32 + mt * 16 + r0;
            const unsigned int* p  = (const unsigned int*)&Ab[row * KB + hh * 8];
            const unsigned int* p2 = (const unsigned int*)&Ab[row * KB + hh * 8 + 16];
#pragma unroll
            for (int i = 0; i < 4; ++i) { af[mt].u[i] = p[i]; af[mt].u[4 + i] = p2[i]; }
        }
#pragma unroll
        for (int nt = 0; nt < 4; ++nt) {
            int col = wn * 64 + nt * 16 + r0;
            Frag16 bf;
            const unsigned int* p  = (const unsigned int*)&Bb[col * KB + hh * 8];
            const unsigned int* p2 = (const unsigned int*)&Bb[col * KB + hh * 8 + 16];
#pragma unroll
            for (int i = 0; i < 4; ++i) { bf.u[i] = p[i]; bf.u[4 + i] = p2[i]; }
#pragma unroll
            for (int mt = 0; mt < 2; ++mt)
                acc[mt][nt] = wmma_bf16(af[mt], bf, acc[mt][nt]);
        }

        if (it + 1 < nsteps)
            stage_store(As[cur ^ 1], Bs[cur ^ 1], t, sr);
        __syncthreads();
    }

    // ---- epilogue ----
#pragma unroll
    for (int nt = 0; nt < 4; ++nt) {
        int col = bn + wn * 64 + nt * 16 + r0;
        float bv = bias[col];
#pragma unroll
        for (int mt = 0; mt < 2; ++mt) {
#pragma unroll
            for (int j = 0; j < 8; ++j) {
                int row = bm + wm * 32 + mt * 16 + hh * 8 + j;
                size_t idx = (size_t)row * N + col;
                float val = acc[mt][nt][j] + bv;
                if (res)  val += res[idx];
                if (relu) val = fmaxf(val, 0.f);
                if (out_mode == 0) {
                    ((float*)Cout)[idx] = val;
                } else if (out_mode == 1) {
                    ((unsigned short*)Cout)[idx] = f2bf(val);
                } else {   // V^T: [B, H, dk, S]
                    int bb = row >> 11, ss = row & (SS - 1);
                    int hd = col >> 6,  dk = col & 63;
                    size_t ti = ((size_t)(bb * HH_ + hd) * 64 + dk) * SS + ss;
                    ((unsigned short*)Cout)[ti] = f2bf(val);
                }
            }
        }
    }
}

// =====================================================================
// Flash attention over bf16 Q/K/V (V pre-transposed). One WG per
// (b, head, 64 q-rows); 4 waves x 16 q-rows. Async LDS staging if avail.
// =====================================================================
__global__ __launch_bounds__(128) void flash_attn_kernel(
    const unsigned short* __restrict__ q16, const unsigned short* __restrict__ k16,
    const unsigned short* __restrict__ vT16, const int* __restrict__ mask,
    float* __restrict__ out)
{
    const int qb   = blockIdx.x * 64;
    const int head = blockIdx.y;
    const int b    = blockIdx.z;
    const int t    = threadIdx.x;
    const int wave = t >> 5;
    const int lane = t & 31;
    const int hh   = lane >> 4;
    const int r0   = lane & 15;

    __shared__ unsigned short Qs[64 * 64];      // [row][dim]
    __shared__ unsigned short Ks[64 * 64];      // [key][dim]  B-layout for Q.K^T
    __shared__ unsigned short Vs[64 * 64];      // [dim][key]  B-layout for P.V
    __shared__ unsigned short Ps[4 * 16 * 64];  // per-wave P strip

    // ---- stage Q tile (pure 16B copies) ----
#ifdef USE_ASYNC_LDS
#pragma unroll
    for (int i = 0; i < 4; ++i) {
        int slot = t + i * 128, row = slot >> 3, grp = slot & 7;
        CP16B(&Qs[row * 64 + grp * 8],
              q16 + ((size_t)b * SS + qb + row) * DD + head * 64 + grp * 8);
    }
    CP_WAIT();
#else
    {
        uint4 tq[4];
#pragma unroll
        for (int i = 0; i < 4; ++i) {
            int slot = t + i * 128, row = slot >> 3, grp = slot & 7;
            tq[i] = *(const uint4*)(q16 + ((size_t)b * SS + qb + row) * DD + head * 64 + grp * 8);
        }
#pragma unroll
        for (int i = 0; i < 4; ++i) {
            int slot = t + i * 128, row = slot >> 3, grp = slot & 7;
            *(uint4*)&Qs[row * 64 + grp * 8] = tq[i];
        }
    }
#endif
    __syncthreads();

    Frag16 qa[2];
#pragma unroll
    for (int s = 0; s < 2; ++s) {
        int row = wave * 16 + r0;
        const unsigned int* p  = (const unsigned int*)&Qs[row * 64 + s * 32 + hh * 8];
        const unsigned int* p2 = (const unsigned int*)&Qs[row * 64 + s * 32 + hh * 8 + 16];
#pragma unroll
        for (int i = 0; i < 4; ++i) { qa[s].u[i] = p[i]; qa[s].u[4 + i] = p2[i]; }
    }

    v8f oacc[4];
#pragma unroll
    for (int nt = 0; nt < 4; ++nt)
#pragma unroll
        for (int j = 0; j < 8; ++j) oacc[nt][j] = 0.f;

    float mstat[8], lstat[8];
#pragma unroll
    for (int j = 0; j < 8; ++j) { mstat[j] = -1e30f; lstat[j] = 0.f; }

    for (int jb = 0; jb < SS / 64; ++jb) {
        __syncthreads();
        // ---- stage K[key][dim] and V^T[dim][key] tiles ----
#ifdef USE_ASYNC_LDS
#pragma unroll
        for (int i = 0; i < 4; ++i) {
            int slot = t + i * 128, row = slot >> 3, grp = slot & 7;
            CP16B(&Ks[row * 64 + grp * 8],
                  k16 + ((size_t)b * SS + jb * 64 + row) * DD + head * 64 + grp * 8);
            CP16B(&Vs[row * 64 + grp * 8],
                  vT16 + ((size_t)(b * HH_ + head) * 64 + row) * SS + jb * 64 + grp * 8);
        }
        CP_WAIT();
#else
        {
            uint4 tk[4], tv[4];
#pragma unroll
            for (int i = 0; i < 4; ++i) {
                int slot = t + i * 128, row = slot >> 3, grp = slot & 7;
                tk[i] = *(const uint4*)(k16 + ((size_t)b * SS + jb * 64 + row) * DD + head * 64 + grp * 8);
                tv[i] = *(const uint4*)(vT16 + ((size_t)(b * HH_ + head) * 64 + row) * SS + jb * 64 + grp * 8);
            }
#pragma unroll
            for (int i = 0; i < 4; ++i) {
                int slot = t + i * 128, row = slot >> 3, grp = slot & 7;
                *(uint4*)&Ks[row * 64 + grp * 8] = tk[i];
                *(uint4*)&Vs[row * 64 + grp * 8] = tv[i];
            }
        }
#endif
        __syncthreads();

        // ---- scores = Q K^T ----
        v8f sc[4];
#pragma unroll
        for (int nt = 0; nt < 4; ++nt) {
            v8f a;
#pragma unroll
            for (int j = 0; j < 8; ++j) a[j] = 0.f;
#pragma unroll
            for (int s = 0; s < 2; ++s) {
                Frag16 kb;
                const unsigned int* p  = (const unsigned int*)&Ks[(nt * 16 + r0) * 64 + s * 32 + hh * 8];
                const unsigned int* p2 = (const unsigned int*)&Ks[(nt * 16 + r0) * 64 + s * 32 + hh * 8 + 16];
#pragma unroll
                for (int i = 0; i < 4; ++i) { kb.u[i] = p[i]; kb.u[4 + i] = p2[i]; }
                a = wmma_bf16(qa[s], kb, a);
            }
            sc[nt] = a;
        }

        // ---- scale (1/sqrt(D)=1/32) + mask-replace ----
        const float scale = 0.03125f;
#pragma unroll
        for (int nt = 0; nt < 4; ++nt) {
            int kidx = jb * 64 + nt * 16 + r0;
            int mval = mask[(size_t)b * SS + kidx];
#pragma unroll
            for (int j = 0; j < 8; ++j) {
                float sv = sc[nt][j] * scale;
                sc[nt][j] = (mval == 0) ? -1e9f : sv;
            }
        }

        // ---- row max (xor-shuffle within each 16-lane half) ----
        float rmax[8];
#pragma unroll
        for (int j = 0; j < 8; ++j) {
            float m = sc[0][j];
#pragma unroll
            for (int nt = 1; nt < 4; ++nt) m = fmaxf(m, sc[nt][j]);
            rmax[j] = m;
        }
#pragma unroll
        for (int d = 1; d < 16; d <<= 1)
#pragma unroll
            for (int j = 0; j < 8; ++j)
                rmax[j] = fmaxf(rmax[j], __shfl_xor(rmax[j], d, 32));

        float alpha[8];
#pragma unroll
        for (int j = 0; j < 8; ++j) {
            float mn = fmaxf(mstat[j], rmax[j]);
            alpha[j] = expf(mstat[j] - mn);
            mstat[j] = mn;
        }

        float rsum[8];
#pragma unroll
        for (int j = 0; j < 8; ++j) rsum[j] = 0.f;
#pragma unroll
        for (int nt = 0; nt < 4; ++nt)
#pragma unroll
            for (int j = 0; j < 8; ++j) {
                float pv = expf(sc[nt][j] - mstat[j]);
                sc[nt][j] = pv;
                rsum[j] += pv;
            }
#pragma unroll
        for (int d = 1; d < 16; d <<= 1)
#pragma unroll
            for (int j = 0; j < 8; ++j)
                rsum[j] += __shfl_xor(rsum[j], d, 32);
#pragma unroll
        for (int j = 0; j < 8; ++j) lstat[j] = lstat[j] * alpha[j] + rsum[j];
#pragma unroll
        for (int nt = 0; nt < 4; ++nt)
#pragma unroll
            for (int j = 0; j < 8; ++j) oacc[nt][j] *= alpha[j];

        // ---- P: C-layout -> A-layout via per-wave LDS strip ----
        unsigned short* Pw = &Ps[wave * 16 * 64];
#pragma unroll
        for (int nt = 0; nt < 4; ++nt)
#pragma unroll
            for (int j = 0; j < 8; ++j)
                Pw[(hh * 8 + j) * 64 + nt * 16 + r0] = f2bf(sc[nt][j]);

        Frag16 pa[2];
#pragma unroll
        for (int s = 0; s < 2; ++s) {
            const unsigned int* p  = (const unsigned int*)&Pw[r0 * 64 + s * 32 + hh * 8];
            const unsigned int* p2 = (const unsigned int*)&Pw[r0 * 64 + s * 32 + hh * 8 + 16];
#pragma unroll
            for (int i = 0; i < 4; ++i) { pa[s].u[i] = p[i]; pa[s].u[4 + i] = p2[i]; }
        }

        // ---- oacc += P V ----
#pragma unroll
        for (int nt = 0; nt < 4; ++nt) {
#pragma unroll
            for (int s = 0; s < 2; ++s) {
                Frag16 vb;
                const unsigned int* p  = (const unsigned int*)&Vs[(nt * 16 + r0) * 64 + s * 32 + hh * 8];
                const unsigned int* p2 = (const unsigned int*)&Vs[(nt * 16 + r0) * 64 + s * 32 + hh * 8 + 16];
#pragma unroll
                for (int i = 0; i < 4; ++i) { vb.u[i] = p[i]; vb.u[4 + i] = p2[i]; }
                oacc[nt] = wmma_bf16(pa[s], vb, oacc[nt]);
            }
        }
    }

    // ---- out[b][q][head*64+dim] = o / l (f32) ----
#pragma unroll
    for (int nt = 0; nt < 4; ++nt)
#pragma unroll
        for (int j = 0; j < 8; ++j) {
            int row = qb + wave * 16 + hh * 8 + j;
            int dim = nt * 16 + r0;
            out[((size_t)b * SS + row) * DD + head * 64 + dim] = oacc[nt][j] / lstat[j];
        }
}

// =====================================================================
// LayerNorm over last dim (D=1024), one row per block (256 threads)
// =====================================================================
__global__ __launch_bounds__(256) void layernorm_kernel(
    const float* __restrict__ x, const float* __restrict__ g,
    const float* __restrict__ be, float* __restrict__ out)
{
    __shared__ float sbuf[8], s2buf[8];
    const int row = blockIdx.x;
    const float* xr = x + (size_t)row * DD;
    float s = 0.f, s2 = 0.f;
    for (int i = threadIdx.x; i < DD; i += 256) {
        float v = xr[i]; s += v; s2 += v * v;
    }
#pragma unroll
    for (int d = 1; d < 32; d <<= 1) {
        s  += __shfl_xor(s,  d, 32);
        s2 += __shfl_xor(s2, d, 32);
    }
    int wave = threadIdx.x >> 5, lane = threadIdx.x & 31;
    if (lane == 0) { sbuf[wave] = s; s2buf[wave] = s2; }
    __syncthreads();
    s = 0.f; s2 = 0.f;
#pragma unroll
    for (int i = 0; i < 8; ++i) { s += sbuf[i]; s2 += s2buf[i]; }
    float mean = s * (1.f / DD);
    float var  = s2 * (1.f / DD) - mean * mean;
    float inv  = rsqrtf(var + 1e-5f);
    for (int i = threadIdx.x; i < DD; i += 256)
        out[(size_t)row * DD + i] = (xr[i] - mean) * inv * g[i] + be[i];
}

// =====================================================================
// Host-side orchestration
// =====================================================================
static inline void launch_gemm(const float* A, const float* W, const float* bias,
                               const float* res, void* C, int M, int N, int K,
                               int relu, int out_mode, hipStream_t stream)
{
    dim3 grid(N / TN, M / TM);
    gemm_bf16_kernel<<<grid, 256, 0, stream>>>(A, W, bias, res, C, M, N, K, relu, out_mode);
}

extern "C" void kernel_launch(void* const* d_in, const int* in_sizes, int n_in,
                              void* d_out, int out_size, void* d_ws, size_t ws_size,
                              hipStream_t stream) {
    (void)in_sizes; (void)n_in; (void)out_size; (void)ws_size;

    const int*   src  = (const int*)d_in[0];
    const int*   mask = (const int*)d_in[1];
    const float* emb  = (const float*)d_in[2];
    const float* Wq   = (const float*)d_in[3];
    const float* bq   = (const float*)d_in[4];
    const float* Wk   = (const float*)d_in[5];
    const float* bk   = (const float*)d_in[6];
    const float* Wv   = (const float*)d_in[7];
    const float* bv   = (const float*)d_in[8];
    const float* Wo   = (const float*)d_in[9];
    const float* bo   = (const float*)d_in[10];
    const float* W1   = (const float*)d_in[11];
    const float* b1   = (const float*)d_in[12];
    const float* W2   = (const float*)d_in[13];
    const float* b2   = (const float*)d_in[14];
    const float* g1   = (const float*)d_in[15];
    const float* be1  = (const float*)d_in[16];
    const float* g2   = (const float*)d_in[17];
    const float* be2  = (const float*)d_in[18];

    float* ws = (float*)d_ws;
    const size_t X = (size_t)MR * DD;   // 4,194,304 floats
    float* x    = ws + 0 * X;
    float* x2   = ws + 1 * X;
    float* attn = ws + 2 * X;
    float* y    = ws + 3 * X;
    float* hff  = ws + 4 * X;                         // 4X floats
    unsigned short* q16  = (unsigned short*)(ws + 8 * X);          // X ushorts
    unsigned short* k16  = q16 + X;
    unsigned short* vT16 = k16 + X;
    // total workspace: 9.5X floats = 152 MB

    embed_pe_kernel<<<(BB * SS * DD) / 256, 256, 0, stream>>>(src, emb, x);

    for (int i = 0; i < LL; ++i) {
        const float* Wq_i = Wq + (size_t)i * DD * DD;
        const float* Wk_i = Wk + (size_t)i * DD * DD;
        const float* Wv_i = Wv + (size_t)i * DD * DD;
        const float* Wo_i = Wo + (size_t)i * DD * DD;
        const float* W1_i = W1 + (size_t)i * DD * FF;
        const float* W2_i = W2 + (size_t)i * FF * DD;

        // Q, K (bf16 row-major), V (bf16 transposed [B,H,dk,S])
        launch_gemm(x, Wq_i, bq + (size_t)i * DD, nullptr, q16,  MR, DD, DD, 0, 1, stream);
        launch_gemm(x, Wk_i, bk + (size_t)i * DD, nullptr, k16,  MR, DD, DD, 0, 1, stream);
        launch_gemm(x, Wv_i, bv + (size_t)i * DD, nullptr, vT16, MR, DD, DD, 0, 2, stream);

        flash_attn_kernel<<<dim3(SS / 64, HH_, BB), 128, 0, stream>>>(q16, k16, vT16, mask, attn);

        launch_gemm(attn, Wo_i, bo + (size_t)i * DD, x, y, MR, DD, DD, 0, 0, stream);
        layernorm_kernel<<<MR, 256, 0, stream>>>(y, g1 + (size_t)i * DD, be1 + (size_t)i * DD, x2);

        launch_gemm(x2, W1_i, b1 + (size_t)i * FF, nullptr, hff, MR, FF, DD, 1, 0, stream);
        launch_gemm(hff, W2_i, b2 + (size_t)i * DD, x2, y, MR, DD, FF, 0, 0, stream);

        float* dst = (i == LL - 1) ? (float*)d_out : x;
        layernorm_kernel<<<MR, 256, 0, stream>>>(y, g2 + (size_t)i * DD, be2 + (size_t)i * DD, dst);
    }
}
